// MicroLLM_41308995453154
// MI455X (gfx1250) — matrix operand order repaired and checked by reading, not compiled, not run on hardware
//
#include <hip/hip_runtime.h>
#include <cstdint>
#include <cstddef>

// ---------------- model dims (match reference) ----------------
#define V_  32000
#define D_  768
#define H_  12
#define L_  8
#define T_  1024
#define B_  4
#define HD_ 64
#define M_  (B_*T_)        // 4096 token rows
#define TD_ (3*D_)         // 2304

// Use the Tensor Data Mover for GEMM A-tile staging (flip to 0 -> async path).
#ifndef USE_TDM
#define USE_TDM 1
#endif

typedef __bf16 bf16;
typedef __attribute__((ext_vector_type(16))) __bf16 v16bf;
typedef __attribute__((ext_vector_type(8)))  float  v8f;

struct alignas(16) bfx8 { bf16 v[8]; };

__device__ inline unsigned short bf_bits(bf16 x) {
  union { bf16 b; unsigned short u; } cv; cv.b = x; return cv.u;
}

// Low 32 bits of a flat pointer to LDS == wave-relative LDS byte address
__device__ inline unsigned lds_addr32(const void* p) {
  return (unsigned)(uintptr_t)p;
}

// CDNA5 async global->LDS copy (VGLOBAL encoding, tracked by ASYNCcnt).
__device__ inline void async_b128(unsigned lds_off, const void* gaddr) {
  asm volatile("global_load_async_to_lds_b128 %0, %1, off"
               :: "v"(lds_off), "v"(gaddr) : "memory");
}
__device__ inline void wait_async0() {
  asm volatile("s_wait_asynccnt 0" ::: "memory");
}
__device__ inline void wait_async1() {
  asm volatile("s_wait_asynccnt 1" ::: "memory");
}
__device__ inline void wait_async8() {
  asm volatile("s_wait_asynccnt 8" ::: "memory");
}

#if USE_TDM
typedef __attribute__((ext_vector_type(4))) unsigned int su4;
typedef __attribute__((ext_vector_type(8))) int si8;
typedef __attribute__((ext_vector_type(4))) int si4;

// Issue a 2D TDM tile load: tile_rows x tile_cols bf16 elements from a
// row-major tensor (row stride = row_elems) into contiguous LDS.
// D# fields per CDNA5 ISA 8.3/8.4 (count=1, type=2, data_size=1 -> 2 bytes).
__device__ inline void tdm_load_2d(const void* gaddr, unsigned lds_byte_addr,
                                   int rows, int row_elems,
                                   int tile_rows, int tile_cols) {
  const unsigned long long ga = (unsigned long long)(uintptr_t)gaddr;
  su4 g0;
  g0[0] = 1u;                                   // count=1 (valid descriptor)
  g0[1] = lds_byte_addr;                        // lds_addr
  g0[2] = (unsigned)(ga & 0xffffffffu);         // global_addr[31:0]
  g0[3] = (unsigned)((ga >> 32) & 0x01ffffffu)  // global_addr[56:32]
          | 0x80000000u;                        // type=2 ("image")
  si8 g1;
  g1[0] = 0x00010000;                           // data_size=1 (2B elements)
  g1[1] = (int)(((unsigned)row_elems & 0xffffu) << 16);        // dim0 lo16
  g1[2] = (int)((((unsigned)row_elems >> 16) & 0xffffu)
          | (((unsigned)rows & 0xffffu) << 16));               // dim0 hi | dim1 lo
  g1[3] = (int)((((unsigned)rows >> 16) & 0xffffu)
          | ((unsigned)tile_cols << 16));                      // dim1 hi | tile_dim0
  g1[4] = (int)(unsigned)tile_rows;                            // tile_dim1 (dim2=0)
  g1[5] = (int)(unsigned)row_elems;                            // dim0_stride[31:0]
  g1[6] = 0;                                                   // stride hi / dim1_stride
  g1[7] = 0;
  const si4 gz4 = {0, 0, 0, 0};
  const si8 gz8 = {0, 0, 0, 0, 0, 0, 0, 0};
  // 6-arg toolchain variant (clang-23 / therock HIP headers)
  __builtin_amdgcn_tensor_load_to_lds(g0, g1, gz4, gz4, gz8, 0);
}
#endif

// Build a v16bf WMMA operand from two contiguous groups of 8 bf16 in LDS.
__device__ inline v16bf make_frag(const bf16* lo, const bf16* hi) {
  const bfx8 a = *(const bfx8*)lo;
  const bfx8 b = *(const bfx8*)hi;
  v16bf f;
#pragma unroll
  for (int i = 0; i < 8; ++i) { f[i] = a.v[i]; f[i + 8] = b.v[i]; }
  return f;
}

__device__ inline v8f wmma_bf16(v16bf a, v16bf b, v8f c) {
  // D = A(16x32 bf16) * B(32x16 bf16) + C(16x16 f32)
  return __builtin_amdgcn_wmma_f32_16x16x32_bf16(false, a, false, b,
                                                 (short)0, c, false, false);
}

// ---------------- fp32 -> bf16 convert (grid-stride) ----------------
__global__ void f32_to_bf16_k(const float* __restrict__ src,
                              bf16* __restrict__ dst, size_t n) {
  size_t i = (size_t)blockIdx.x * blockDim.x + threadIdx.x;
  size_t stride = (size_t)gridDim.x * blockDim.x;
  for (; i < n; i += stride) dst[i] = (bf16)src[i];
}

// ---------------- embedding: x = wte[idx] + wpe[t] ----------------
__global__ void embed_k(const int* __restrict__ idx,
                        const float* __restrict__ wte,
                        const float* __restrict__ wpe,
                        float* __restrict__ x) {
  const int row = blockIdx.x;          // 0..M_-1
  const int t   = row % T_;
  const int tok = idx[row];
  const float* we = wte + (size_t)tok * D_;
  const float* pe = wpe + (size_t)t * D_;
  float* xr = x + (size_t)row * D_;
  for (int d = threadIdx.x; d < D_; d += blockDim.x)
    xr[d] = we[d] + pe[d];
}

// ---------------- layernorm (f32 in, bf16 out), wave per row ----------------
__global__ __launch_bounds__(256)
void layernorm_bf16_k(const float* __restrict__ x,
                      const float* __restrict__ g,
                      const float* __restrict__ be,
                      bf16* __restrict__ out) {
  const int row  = blockIdx.x * 8 + (threadIdx.x >> 5);
  const int lane = threadIdx.x & 31;
  const float* xr = x + (size_t)row * D_;
  float v[D_ / 32];
  float s = 0.f, s2 = 0.f;
#pragma unroll
  for (int i = 0; i < D_ / 32; ++i) {
    v[i] = xr[lane + 32 * i];
    s += v[i]; s2 += v[i] * v[i];
  }
#pragma unroll
  for (int off = 16; off >= 1; off >>= 1) {
    s  += __shfl_xor(s,  off);
    s2 += __shfl_xor(s2, off);
  }
  const float mu   = s / (float)D_;
  const float var  = s2 / (float)D_ - mu * mu;
  const float rstd = rsqrtf(var + 1e-5f);
  bf16* orow = out + (size_t)row * D_;
#pragma unroll
  for (int i = 0; i < D_ / 32; ++i) {
    const int d = lane + 32 * i;
    orow[d] = (bf16)((v[i] - mu) * rstd * g[d] + be[d]);
  }
}

// ---------------- WMMA GEMM: C[M,N] = A[M,K](bf16) @ B[K,N](bf16) ----------
// Block tile 64x256, 8 waves (2Mx4N), wave tile 32x64 -> 8 WMMA per K-step.
// Double-buffered 2-deep pipeline: A tile via TDM (or async-LDS), B tile via
// regs one stage ahead; partial counter waits keep next stage in flight.
// flags: 1=bias, 2=gelu(exact erf), 4=residual add into `resid`,
//        8=write f32 to outf, 16=write bf16 to outbf
// Requirements (all satisfied here): M%64==0, N%256==0, K%32==0, K/32>=2.
template <int FLAGS>
__global__ __launch_bounds__(256)
void gemm_wmma_k(const bf16* __restrict__ A, const bf16* __restrict__ Bw,
                 const float* __restrict__ bias, float* __restrict__ resid,
                 float* __restrict__ outf, bf16* __restrict__ outbf,
                 int M, int N, int K) {
  constexpr bool HAS_BIAS = FLAGS & 1;
  constexpr bool DO_GELU  = FLAGS & 2;
  constexpr bool RESID    = FLAGS & 4;
  constexpr bool OUTF32   = FLAGS & 8;
  constexpr bool OUTBF16  = FLAGS & 16;

  __shared__ bf16 lds_a[2][64 * 32];      // A tiles row-major [row][k]
  __shared__ bf16 lds_bt[2][256 * 32];    // B tiles transposed [n][k]

  const int tid  = threadIdx.x;
  const int lane = tid & 31;
  const int wave = tid >> 5;
  const int wm = wave >> 2;            // 0..1
  const int wn = wave & 3;             // 0..3
  const int mBase = blockIdx.y * 64;
  const int nBase = blockIdx.x * 256;

  const int a_row = tid >> 2;              // 0..63
  const int a_col = (tid & 3) * 8;         // 0,8,16,24
  const int b_k = (tid & 15) * 2;          // 0..30 even
  const int b_n = (tid >> 4) * 16;         // 0..240

  const int kb   = (lane < 16) ? 0 : 8;    // A K-base per half-wave
  const int kb2  = (lane < 16) ? 0 : 16;   // unused (B fragments use kb too)
  const int ln16 = lane & 15;
  const int nT   = K >> 5;

  struct Bregs { bfx8 r[4]; };
  auto loadB = [&](int t) {
    Bregs rb;
    const bf16* brow0 = Bw + (size_t)(t * 32 + b_k) * N + nBase + b_n;
    rb.r[0] = *(const bfx8*)(brow0);
    rb.r[1] = *(const bfx8*)(brow0 + 8);
    rb.r[2] = *(const bfx8*)(brow0 + N);
    rb.r[3] = *(const bfx8*)(brow0 + N + 8);
    return rb;
  };
  auto storeB = [&](const Bregs& rb, int buf) {
    unsigned* bt32 = (unsigned*)&lds_bt[buf][0];
#pragma unroll
    for (int e = 0; e < 8; ++e) {
      bt32[(b_n + e) * 16 + (b_k >> 1)] =
          (unsigned)bf_bits(rb.r[0].v[e]) | ((unsigned)bf_bits(rb.r[2].v[e]) << 16);
      bt32[(b_n + 8 + e) * 16 + (b_k >> 1)] =
          (unsigned)bf_bits(rb.r[1].v[e]) | ((unsigned)bf_bits(rb.r[3].v[e]) << 16);
    }
  };
  auto issueA = [&](int t, int buf) {
#if USE_TDM
    if (wave == 0)
      tdm_load_2d(A + (size_t)mBase * K + t * 32, lds_addr32(&lds_a[buf][0]),
                  M, K, /*tile_rows=*/64, /*tile_cols=*/32);
#else
    async_b128(lds_addr32(&lds_a[buf][a_row * 32 + a_col]),
               A + (size_t)(mBase + a_row) * K + t * 32 + a_col);
#endif
  };
  auto waitA = [&](bool pending) {
#if USE_TDM
    if (pending) __builtin_amdgcn_s_wait_tensorcnt(1);
    else         __builtin_amdgcn_s_wait_tensorcnt(0);
#else
    if (pending) wait_async1(); else wait_async0();
#endif
  };

  v8f acc[2][4] = {};

  // pipeline prologue: stage 0 in LDS, stage 1 A in flight, stage 1 B in regs
  Bregs brg = loadB(0);
  issueA(0, 0);
  storeB(brg, 0);
  if (nT > 1) { brg = loadB(1); issueA(1, 1); }
  waitA(nT > 1);
  __syncthreads();

  for (int t = 0; t < nT; ++t) {
    const int buf = t & 1;
    const bf16* la = &lds_a[buf][0];
    const bf16* lb = &lds_bt[buf][0];

    v16bf af[2], bfr[4];
#pragma unroll
    for (int sm = 0; sm < 2; ++sm) {
      const int m = wm * 32 + sm * 16 + ln16;
      af[sm] = make_frag(la + m * 32 + kb, la + m * 32 + kb + 16);
    }
#pragma unroll
    for (int sn = 0; sn < 4; ++sn) {
      const int n = wn * 64 + sn * 16 + ln16;
      bfr[sn] = make_frag(lb + n * 32 + kb, lb + n * 32 + kb + 16);
    }
#pragma unroll
    for (int sm = 0; sm < 2; ++sm)
#pragma unroll
      for (int sn = 0; sn < 4; ++sn)
        acc[sm][sn] = wmma_bf16(af[sm], bfr[sn], acc[sm][sn]);

    __syncthreads();                       // all waves done reading buf
    if (t + 1 < nT) {
      storeB(brg, (t + 1) & 1);            // B tile t+1 into LDS
      const bool more = (t + 2 < nT);
      if (more) {
        issueA(t + 2, buf);                // overlaps compute of t+1
        __builtin_prefetch(Bw + (size_t)((t + 2) * 32 + b_k) * N + nBase + b_n, 0, 0);
        brg = loadB(t + 2);
      }
      waitA(more);                         // A tile t+1 resident
      __syncthreads();                     // stage t+1 visible to all waves
    }
  }

  // epilogue: C layout = lane holds col N=lane%16, rows r + (lane<16?0:8)
  const int hi8 = (lane < 16) ? 0 : 8;
#pragma unroll
  for (int sm = 0; sm < 2; ++sm) {
#pragma unroll
    for (int sn = 0; sn < 4; ++sn) {
      const int nG = nBase + wn * 64 + sn * 16 + ln16;
#pragma unroll
      for (int r = 0; r < 8; ++r) {
        const int mG = mBase + wm * 32 + sm * 16 + hi8 + r;
        float val = acc[sm][sn][r];
        if (HAS_BIAS) val += bias[nG];
        if (DO_GELU)  val = 0.5f * val * (1.0f + erff(val * 0.70710678118654752f));
        const size_t o = (size_t)mG * (size_t)N + (size_t)nG;
        if (RESID) { const float t2 = resid[o] + val; resid[o] = t2; val = t2; }
        if (OUTF32)  outf[o]  = val;
        if (OUTBF16) outbf[o] = (bf16)val;
      }
    }
  }
}

// ---------------- flash attention with WMMA ----------------
// qkv: [M_, 3D] bf16 (q|k|v per head, HD=64). ybf: [M_, D] bf16.
// grid (B*H, 16), block 128 (=4 waves); wave handles one 16-row Q tile.
// Per-wave double-buffered K/V chunks; async K copies pipelined one chunk
// ahead with partial ASYNCcnt waits (no barriers: buffers are wave-private).
__global__ __launch_bounds__(128)
void attn_wmma_k(const bf16* __restrict__ qkv, bf16* __restrict__ ybf) {
  __shared__ bf16 smem[4 * 9728]; // per wave: Q 1024 | 2xK 2048 | 2xV^T 2048 | P 512
  const int tid  = threadIdx.x;
  const int lane = tid & 31;
  const int wave = tid >> 5;
  const int b = blockIdx.x / H_;
  const int h = blockIdx.x % H_;
  const int qt = blockIdx.y * 4 + wave;      // 0..63

  bf16* ldsQ  = smem + wave * 9728;
  bf16* ldsK  = ldsQ + 1024;       // [2][2048]
  bf16* ldsVT = ldsK + 4096;       // [2][2048]
  bf16* ldsP  = ldsVT + 4096;

  const size_t rowQ = (size_t)(b * T_ + qt * 16);
  const int qoff = h * HD_;
  const int koff = D_ + h * HD_;
  const int voff = 2 * D_ + h * HD_;

  const int kb   = (lane < 16) ? 0 : 8;
  const int kb2  = (lane < 16) ? 0 : 16;
  const int ln16 = lane & 15;
  const int hi8  = (lane < 16) ? 0 : 8;

  auto stageKV = [&](int kt, int buf) {
    const size_t rowK = (size_t)(b * T_ + kt * 32);
    bf16* K_  = ldsK  + buf * 2048;
    bf16* VT_ = ldsVT + buf * 2048;
    for (int i = lane; i < 256; i += 32) {            // 8 async b128 per lane-loop
      const int r = i >> 3, c = (i & 7) * 8;
      async_b128(lds_addr32(K_ + r * 64 + c),
                 qkv + (rowK + r) * TD_ + koff + c);
    }
    for (int i = lane; i < 2048; i += 32) {           // V transposed [hd][kpos]
      const int r = i >> 6, c = i & 63;
      VT_[c * 32 + r] = qkv[(rowK + r) * TD_ + voff + c];
    }
  };

  // stage Q tile (16x64 bf16) via async copies (wave-private region)
  for (int i = lane; i < 128; i += 32) {
    const int r = i >> 3, c = (i & 7) * 8;
    async_b128(lds_addr32(ldsQ + r * 64 + c),
               qkv + (rowQ + r) * TD_ + qoff + c);
  }
  const int nch = (qt * 16 + 47) / 32;       // causal: k chunks of 32
  stageKV(0, 0);
  wait_async8();                             // Q resident (async done in order)

  v16bf aQ[2];
#pragma unroll
  for (int c = 0; c < 2; ++c) {
    const bf16* base = ldsQ + ln16 * 64 + c * 32;
    aQ[c] = make_frag(base + kb, base + kb + 16);
  }

  v8f o[4] = {};
  float mrow[8], lrow[8];
#pragma unroll
  for (int r = 0; r < 8; ++r) { mrow[r] = -3.0e38f; lrow[r] = 0.f; }

  for (int kt = 0; kt < nch; ++kt) {
    const int buf = kt & 1;
    if (kt + 1 < nch) { stageKV(kt + 1, buf ^ 1); wait_async8(); }
    else              { wait_async0(); }
    const bf16* K_  = ldsK  + buf * 2048;
    const bf16* VT_ = ldsVT + buf * 2048;

    // S = Q K^T  (two 16-col sub-tiles, hd split 2x32)
    v8f s[2] = {};
#pragma unroll
    for (int sk = 0; sk < 2; ++sk)
#pragma unroll
      for (int c = 0; c < 2; ++c) {
        const bf16* base = K_ + (sk * 16 + ln16) * 64 + c * 32 + kb;
        s[sk] = wmma_bf16(aQ[c], make_frag(base, base + 16), s[sk]);
      }

    // online softmax (branchless mask; row spread over 16 lanes)
    const int j0 = kt * 32 + ln16;
#pragma unroll
    for (int r = 0; r < 8; ++r) {
      const int mG = qt * 16 + hi8 + r;
      float s0 = s[0][r] * 0.125f;
      float s1 = s[1][r] * 0.125f;
      if (j0 > mG)      s0 = -3.0e38f;
      if (j0 + 16 > mG) s1 = -3.0e38f;
      float mx = fmaxf(s0, s1);
#pragma unroll
      for (int off = 8; off >= 1; off >>= 1)
        mx = fmaxf(mx, __shfl_xor(mx, off, 16));
      const float nm    = fmaxf(mrow[r], mx);
      const float alpha = __expf(mrow[r] - nm);
      const float p0 = __expf(s0 - nm);
      const float p1 = __expf(s1 - nm);
      float rs = p0 + p1;
#pragma unroll
      for (int off = 8; off >= 1; off >>= 1)
        rs += __shfl_xor(rs, off, 16);
      lrow[r] = lrow[r] * alpha + rs;
      mrow[r] = nm;
#pragma unroll
      for (int c2 = 0; c2 < 4; ++c2) o[c2][r] *= alpha;
      // C-layout -> A-layout transpose via wave-private LDS
      ldsP[(hi8 + r) * 32 + ln16]      = (bf16)p0;
      ldsP[(hi8 + r) * 32 + 16 + ln16] = (bf16)p1;
    }

    // O += P V
    const bf16* pb = ldsP + ln16 * 32;
    const v16bf aP = make_frag(pb + kb, pb + kb + 16);
#pragma unroll
    for (int c2 = 0; c2 < 4; ++c2) {
      const bf16* vb = VT_ + (c2 * 16 + ln16) * 32 + kb;
      o[c2] = wmma_bf16(aP, make_frag(vb, vb + 16), o[c2]);
    }
  }

#pragma unroll
  for (int c2 = 0; c2 < 4; ++c2)
#pragma unroll
    for (int r = 0; r < 8; ++r) {
      const float vout = o[c2][r] / lrow[r];
      ybf[(rowQ + hi8 + r) * D_ + h * HD_ + c2 * 16 + ln16] = (bf16)vout;
    }
}

// ---------------- per-row NLL over vocab ----------------
__global__ __launch_bounds__(256)
void nll_row_k(const float* __restrict__ logits, const int* __restrict__ targets,
               float* __restrict__ nll) {
  __shared__ float red[256];
  const int row = blockIdx.x;
  const float* lr = logits + (size_t)row * V_;
  float mx = -3.0e38f;
  for (int i = threadIdx.x; i < V_; i += 256) mx = fmaxf(mx, lr[i]);
  red[threadIdx.x] = mx; __syncthreads();
  for (int s = 128; s >= 1; s >>= 1) {
    if (threadIdx.x < s) red[threadIdx.x] = fmaxf(red[threadIdx.x], red[threadIdx.x + s]);
    __syncthreads();
  }
  mx = red[0]; __syncthreads();
  float se = 0.f;
  for (int i = threadIdx.x; i < V_; i += 256) se += __expf(lr[i] - mx);
  red[threadIdx.x] = se; __syncthreads();
  for (int s = 128; s >= 1; s >>= 1) {
    if (threadIdx.x < s) red[threadIdx.x] += red[threadIdx.x + s];
    __syncthreads();
  }
  if (threadIdx.x == 0) {
    const float lse = mx + __logf(red[0]);
    nll[row] = lse - lr[targets[row]];
  }
}

__global__ __launch_bounds__(256)
void mean_k(const float* __restrict__ nll, float* __restrict__ out) {
  __shared__ float red[256];
  float s = 0.f;
  for (int i = threadIdx.x; i < M_; i += 256) s += nll[i];
  red[threadIdx.x] = s; __syncthreads();
  for (int so = 128; so >= 1; so >>= 1) {
    if (threadIdx.x < so) red[threadIdx.x] += red[threadIdx.x + so];
    __syncthreads();
  }
  if (threadIdx.x == 0) out[0] = red[0] / (float)M_;
}

// ---------------- host orchestration ----------------
extern "C" void kernel_launch(void* const* d_in, const int* in_sizes, int n_in,
                              void* d_out, int out_size, void* d_ws, size_t ws_size,
                              hipStream_t stream) {
  (void)in_sizes; (void)n_in; (void)out_size; (void)ws_size;
  const int*   idx     = (const int*)  d_in[0];
  const int*   targets = (const int*)  d_in[1];
  const float* wte     = (const float*)d_in[2];
  const float* wpe     = (const float*)d_in[3];
  const float* ln1_g   = (const float*)d_in[4];
  const float* ln1_b   = (const float*)d_in[5];
  const float* w_attn  = (const float*)d_in[6];
  const float* w_proj  = (const float*)d_in[7];
  const float* ln2_g   = (const float*)d_in[8];
  const float* ln2_b   = (const float*)d_in[9];
  const float* w_fc    = (const float*)d_in[10];
  const float* b_fc    = (const float*)d_in[11];
  const float* w_out   = (const float*)d_in[12];
  const float* b_out   = (const float*)d_in[13];
  const float* lnf_g   = (const float*)d_in[14];
  const float* lnf_b   = (const float*)d_in[15];
  const float* w_head  = (const float*)d_in[16];

  float* logits = (float*)d_out;                       // [M_, V_]
  float* loss   = logits + (size_t)M_ * V_;            // 1 elem

  // workspace carve (256B aligned)
  char* p = (char*)d_ws;
  auto carve = [&](size_t bytes) -> void* {
    void* r = (void*)p; p += (bytes + 255) & ~(size_t)255; return r;
  };
  const size_t n_wattn = (size_t)L_ * D_ * TD_;
  const size_t n_wproj = (size_t)L_ * D_ * D_;
  const size_t n_wfc   = (size_t)L_ * D_ * 4 * D_;
  const size_t n_wout  = (size_t)L_ * 4 * D_ * D_;
  const size_t n_whead = (size_t)D_ * V_;
  bf16*  wattn_bf = (bf16*)carve(n_wattn * 2);
  bf16*  wproj_bf = (bf16*)carve(n_wproj * 2);
  bf16*  wfc_bf   = (bf16*)carve(n_wfc   * 2);
  bf16*  wout_bf  = (bf16*)carve(n_wout  * 2);
  bf16*  whead_bf = (bf16*)carve(n_whead * 2);
  float* x     = (float*)carve((size_t)M_ * D_ * 4);     // residual stream
  bf16*  hbf   = (bf16*) carve((size_t)M_ * D_ * 2);     // LN output
  bf16*  qkvbf = (bf16*) carve((size_t)M_ * TD_ * 2);    // qkv (bf16)
  bf16*  ybf   = (bf16*) carve((size_t)M_ * D_ * 2);     // attn output
  bf16*  gbf   = (bf16*) carve((size_t)M_ * 4 * D_ * 2); // gelu(fc) output
  float* nll   = (float*)carve((size_t)M_ * 4);

  // 1) weight conversion fp32 -> bf16
  f32_to_bf16_k<<<4096, 256, 0, stream>>>(w_attn, wattn_bf, n_wattn);
  f32_to_bf16_k<<<4096, 256, 0, stream>>>(w_proj, wproj_bf, n_wproj);
  f32_to_bf16_k<<<4096, 256, 0, stream>>>(w_fc,   wfc_bf,   n_wfc);
  f32_to_bf16_k<<<4096, 256, 0, stream>>>(w_out,  wout_bf,  n_wout);
  f32_to_bf16_k<<<4096, 256, 0, stream>>>(w_head, whead_bf, n_whead);

  // 2) embedding
  embed_k<<<M_, 256, 0, stream>>>(idx, wte, wpe, x);

  const dim3 blk(256);
  const dim3 grid_qkv (TD_    / 256, M_ / 64);
  const dim3 grid_proj(D_     / 256, M_ / 64);
  const dim3 grid_fc  (4 * D_ / 256, M_ / 64);
  const dim3 grid_head(V_     / 256, M_ / 64);

  // 3) transformer layers
  for (int l = 0; l < L_; ++l) {
    const bf16* wA = wattn_bf + (size_t)l * D_ * TD_;
    const bf16* wP = wproj_bf + (size_t)l * D_ * D_;
    const bf16* wF = wfc_bf   + (size_t)l * D_ * 4 * D_;
    const bf16* wO = wout_bf  + (size_t)l * 4 * D_ * D_;
    const float* bF = b_fc  + (size_t)l * 4 * D_;
    const float* bO = b_out + (size_t)l * D_;

    layernorm_bf16_k<<<M_ / 8, blk, 0, stream>>>(x, ln1_g + (size_t)l * D_,
                                                 ln1_b + (size_t)l * D_, hbf);
    // qkv = h @ w_attn                                  (bf16 out)
    gemm_wmma_k<16><<<grid_qkv, blk, 0, stream>>>(hbf, wA, nullptr, nullptr,
                                                  nullptr, qkvbf, M_, TD_, D_);
    // flash attention -> ybf (bf16)
    attn_wmma_k<<<dim3(B_ * H_, 16), dim3(128), 0, stream>>>(qkvbf, ybf);
    // x += y @ w_proj                                   (residual)
    gemm_wmma_k<4><<<grid_proj, blk, 0, stream>>>(ybf, wP, nullptr, x,
                                                  nullptr, nullptr, M_, D_, D_);
    layernorm_bf16_k<<<M_ / 8, blk, 0, stream>>>(x, ln2_g + (size_t)l * D_,
                                                 ln2_b + (size_t)l * D_, hbf);
    // g = gelu(h @ w_fc + b_fc)                         (bias+gelu, bf16 out)
    gemm_wmma_k<1 | 2 | 16><<<grid_fc, blk, 0, stream>>>(hbf, wF, bF, nullptr,
                                                         nullptr, gbf,
                                                         M_, 4 * D_, D_);
    // x += g @ w_out + b_out                            (bias+residual)
    gemm_wmma_k<1 | 4><<<grid_proj, blk, 0, stream>>>(gbf, wO, bO, x,
                                                      nullptr, nullptr,
                                                      M_, D_, 4 * D_);
  }

  // 4) final LN + head
  layernorm_bf16_k<<<M_ / 8, blk, 0, stream>>>(x, lnf_g, lnf_b, hbf);
  gemm_wmma_k<8><<<grid_head, blk, 0, stream>>>(hbf, whead_bf, nullptr, nullptr,
                                                logits, nullptr, M_, V_, D_);

  // 5) loss = mean NLL
  nll_row_k<<<M_, blk, 0, stream>>>(logits, targets, nll);
  mean_k<<<1, blk, 0, stream>>>(nll, loss);
}
